// MambaIVModel_81415400063229
// MI455X (gfx1250) — matrix operand order repaired and checked by reading, not compiled, not run on hardware
//
#include <hip/hip_runtime.h>
#include <hip/hip_bf16.h>
#include <math.h>

// ---------------------------------------------------------------------------
// Mamba forward (B=64, L=257, d_model=768, d_inner=1536, d_state=16, 4 layers)
// GEMMs: bf16 WMMA (v_wmma_f32_16x16x32_bf16), f32 accumulate, with
// GLOBAL_LOAD_ASYNC_TO_LDS_B128 double-buffered staging (ASYNCcnt pipeline).
// Selective scan stays f32 VALU (sequential in L, parallel over B*d_inner).
// ---------------------------------------------------------------------------

typedef __attribute__((ext_vector_type(16))) __bf16 v16bf;
typedef __attribute__((ext_vector_type(8)))  float  v8f;

static constexpr int BATCH  = 64;
static constexpr int SEQ    = 257;           // 1 start token + 256
static constexpr int DMODEL = 768;
static constexpr int DINNER = 1536;
static constexpr int DSTATE = 16;
static constexpr int DTRANK = 48;
static constexpr int DTPAD  = 64;            // dt_rank padded to WMMA K multiple
static constexpr int NLAYERS= 4;
static constexpr int MROWS  = BATCH * SEQ;   // 16448 = 257 * 64

static __device__ __forceinline__ unsigned short f2bf(float f) {
  union { float f; unsigned u; } v; v.f = f;
  unsigned r = v.u + 0x7FFFu + ((v.u >> 16) & 1u);   // round-to-nearest-even
  return (unsigned short)(r >> 16);
}
static __device__ __forceinline__ float siluf(float v) {
  return v / (1.f + __expf(-v));
}

// Async DMA: global -> LDS, 16B per lane, tracked by ASYNCcnt.
static __device__ __forceinline__ void async_copy_b128(void* lds_dst,
                                                       const void* gsrc) {
  unsigned lds = (unsigned)(uintptr_t)lds_dst;            // LDS byte offset
  unsigned long long ga = (unsigned long long)(uintptr_t)gsrc;
  asm volatile("global_load_async_to_lds_b128 %0, %1, off"
               :: "v"(lds), "v"(ga) : "memory");
}
template<int N>
static __device__ __forceinline__ void wait_asynccnt() {
  asm volatile("s_wait_asynccnt %0" :: "n"(N) : "memory");
}

// ---------------------------------------------------------------------------
// WMMA GEMM: C[M,N] (f32) = A[M,K] (bf16 row-major) * W[N,K]^T (bf16 row-major)
// bias (per-column, nullable), act: 0 = none, 1 = softplus.
// BM = WAVES_M*WT_M*16, BN = WAVES_N*WT_N*16, BK = 32.
// Double-buffered LDS, staged with global_load_async_to_lds_b128.
// All shapes used here divide the tile sizes exactly (no bounds checks).
// ---------------------------------------------------------------------------
template<int WAVES_M, int WAVES_N, int WT_M, int WT_N>
__global__ __launch_bounds__(WAVES_M * WAVES_N * 32)
void gemm_bf16_wmma(const unsigned short* __restrict__ A,
                    const unsigned short* __restrict__ W,
                    float* __restrict__ C,
                    int M, int N, int K,
                    const float* __restrict__ bias, int act)
{
  constexpr int BM  = WAVES_M * WT_M * 16;
  constexpr int BN  = WAVES_N * WT_N * 16;
  constexpr int NT  = WAVES_M * WAVES_N * 32;
  constexpr int CHA = BM * 4;                       // 16B chunks in A tile
  constexpr int CHB = BN * 4;                       // 16B chunks in B tile (pow2)
  constexpr int ITA = (CHA + NT - 1) / NT;          // async issues per thread (A)
  constexpr int ITB = (CHB + NT - 1) / NT;          // async issues per thread (B)
  constexpr int ISSUES = ITA + ITB;                 // per-thread issues per tile

  __shared__ uint4 Al[2][CHA];                      // [buf][BM rows x 32 bf16]
  __shared__ uint4 Bl[2][CHB];                      // [buf][BN rows x 32 bf16]

  const int tid  = threadIdx.x & (NT - 1);          // provably < NT: no exec guards
  const int lane = tid & 31;
  const int wid  = tid >> 5;
  const int wm0  = (wid % WAVES_M) * WT_M * 16;
  const int wn0  = (wid / WAVES_M) * WT_N * 16;
  const int bM   = blockIdx.x * BM;
  const int bN   = blockIdx.y * BN;
  const int K8   = K >> 3;

  const uint4* __restrict__ Aq = (const uint4*)A;
  const uint4* __restrict__ Wq = (const uint4*)W;

  // Stage one K-tile (32 wide) into LDS buffer `buf` via async DMA.
  auto stage = [&](int k8, int buf) {
#pragma unroll
    for (int c0 = 0; c0 < ITA * NT; c0 += NT) {
      const int c = (c0 + tid) & (CHA - 1);         // exact when CHA%NT==0
      async_copy_b128(&Al[buf][c],
                      Aq + (size_t)(bM + (c >> 2)) * K8 + k8 + (c & 3));
    }
#pragma unroll
    for (int c0 = 0; c0 < ITB * NT; c0 += NT) {
      const int c = (c0 + tid) & (CHB - 1);         // dup-issue if CHB < NT (benign)
      async_copy_b128(&Bl[buf][c],
                      Wq + (size_t)(bN + (c >> 2)) * K8 + k8 + (c & 3));
    }
  };

  v8f acc[WT_M][WT_N];
  const v8f vzero = {0.f, 0.f, 0.f, 0.f, 0.f, 0.f, 0.f, 0.f};
#pragma unroll
  for (int i = 0; i < WT_M; i++)
#pragma unroll
    for (int j = 0; j < WT_N; j++) acc[i][j] = vzero;

  union Frag { uint4 q[2]; v16bf v; };
  const int hl = (lane < 16) ? 0 : 1;

  auto compute = [&](int buf) {
    // Fragment loads per ISA 7.12.2 lane layouts.
    Frag fa[WT_M], fb[WT_N];
#pragma unroll
    for (int tm = 0; tm < WT_M; tm++) {
      const int r = wm0 + tm * 16 + (lane & 15);
      fa[tm].q[0] = Al[buf][r * 4 + hl];            // K = {0..7 | 8..15}
      fa[tm].q[1] = Al[buf][r * 4 + 2 + hl];        // K = {16..23 | 24..31}
    }
#pragma unroll
    for (int tn = 0; tn < WT_N; tn++) {
      const int r = wn0 + tn * 16 + (lane & 15);
      fb[tn].q[0] = Bl[buf][r * 4 + hl];            // K = {0..7 | 8..15}
      fb[tn].q[1] = Bl[buf][r * 4 + 2 + hl];
    }
#pragma unroll
    for (int tm = 0; tm < WT_M; tm++)
#pragma unroll
      for (int tn = 0; tn < WT_N; tn++)
        acc[tm][tn] = __builtin_amdgcn_wmma_f32_16x16x32_bf16(
            false, fa[tm].v, false, fb[tn].v, (short)0, acc[tm][tn], false, false);
  };

  // Software pipeline over K tiles (double buffer).
  const int KT = K >> 5;
  stage(0, 0);
  int kt = 0;
  for (; kt + 1 < KT; kt++) {
    stage((kt + 1) << 2, (kt + 1) & 1);             // k8 = (kt+1)*4
    wait_asynccnt<ISSUES>();                        // current tile landed
    __syncthreads();
    compute(kt & 1);
    __syncthreads();                                // done reading: refill OK
  }
  wait_asynccnt<0>();
  __syncthreads();
  compute(kt & 1);

  // Epilogue: C/D layout — VGPR r, lane L -> (M = r + 8*(L>=16), N = L%16).
  const int mlo = (lane < 16) ? 0 : 8;
  const int nlo = lane & 15;
#pragma unroll
  for (int tm = 0; tm < WT_M; tm++)
#pragma unroll
    for (int tn = 0; tn < WT_N; tn++) {
      const int col = bN + wn0 + tn * 16 + nlo;
      const float bv = bias ? bias[col] : 0.f;
#pragma unroll
      for (int r = 0; r < 8; r++) {
        const int row = bM + wm0 + tm * 16 + mlo + r;
        float v = acc[tm][tn][r] + bv;
        if (act == 1) v = (v > 20.f) ? v : log1pf(__expf(v));
        C[(size_t)row * N + col] = v;
      }
    }
}

// ---------------------------------------------------------------------------
// cond[b,e] = physical[b,:] . phys_w[e,:] + phys_b[e]
// ---------------------------------------------------------------------------
__global__ void cond_kernel(const float* __restrict__ phys,
                            const float* __restrict__ pw,
                            const float* __restrict__ pb,
                            float* __restrict__ cond)
{
  int i = blockIdx.x * blockDim.x + threadIdx.x;
  if (i >= BATCH * DMODEL) return;
  int b = i / DMODEL, e = i % DMODEL;
  float s = pb[e];
#pragma unroll
  for (int k = 0; k < 16; k++) s += phys[b * 16 + k] * pw[e * 16 + k];
  cond[i] = s;
}

// h[b,t,e] = seq(b,t)*val_w[e] + val_b[e] + cond[b,e]; also bf16 copy.
__global__ void embed_kernel(const float* __restrict__ tgt,
                             const float* __restrict__ vw,
                             const float* __restrict__ vb,
                             const float* __restrict__ start,
                             const float* __restrict__ cond,
                             float* __restrict__ h,
                             unsigned short* __restrict__ hbf)
{
  long long i = (long long)blockIdx.x * blockDim.x + threadIdx.x;
  if (i >= (long long)MROWS * DMODEL) return;
  int e = (int)(i % DMODEL);
  long long m = i / DMODEL;
  int b = (int)(m / SEQ), t = (int)(m % SEQ);
  float sv = (t == 0) ? start[0] : tgt[b * (SEQ - 1) + (t - 1)];
  float v = sv * vw[e] + vb[e] + cond[b * DMODEL + e];
  h[i] = v;
  hbf[i] = f2bf(v);
}

__global__ void f32_to_bf16_kernel(const float* __restrict__ s,
                                   unsigned short* __restrict__ d,
                                   long long n)
{
  long long i = (long long)blockIdx.x * blockDim.x + threadIdx.x;
  if (i < n) d[i] = f2bf(s[i]);
}

// dst[r, 0:kdst] = bf16(src[r*stride + k]) for k<kvalid, else 0  (K padding)
__global__ void padk_bf16_kernel(const float* __restrict__ s,
                                 unsigned short* __restrict__ d,
                                 long long rows, int kvalid, int stride, int kdst)
{
  long long i = (long long)blockIdx.x * blockDim.x + threadIdx.x;
  if (i >= rows * kdst) return;
  int k = (int)(i % kdst);
  long long r = i / kdst;
  d[i] = (k < kvalid) ? f2bf(s[r * stride + k]) : (unsigned short)0;
}

// Depthwise causal conv (width 4) + bias + SiLU. x lives in xz[:, 0:DINNER].
__global__ void conv_silu_kernel(const float* __restrict__ xz,
                                 const float* __restrict__ cw,
                                 const float* __restrict__ cb,
                                 float* __restrict__ xo,
                                 unsigned short* __restrict__ xobf)
{
  long long i = (long long)blockIdx.x * blockDim.x + threadIdx.x;
  if (i >= (long long)MROWS * DINNER) return;
  int d = (int)(i % DINNER);
  long long m = i / DINNER;
  int b = (int)(m / SEQ), t = (int)(m % SEQ);
  const float* w = cw + d * 4;
  float s = cb[d];
#pragma unroll
  for (int k = 0; k < 4; k++) {
    int tt = t - 3 + k;
    if (tt >= 0) s += xz[(size_t)(b * SEQ + tt) * (2 * DINNER) + d] * w[k];
  }
  s = siluf(s);
  xo[i] = s;
  xobf[i] = f2bf(s);
}

// ---------------------------------------------------------------------------
// Selective scan. Block = (batch b, 512-channel chunk). Per thread: one
// d-channel, 16 states in registers. B_t/C_t staged through LDS each step.
// Fused: y = (scan + x*D) * silu(z), emitted directly as bf16 for out_proj.
// ---------------------------------------------------------------------------
__global__ __launch_bounds__(512)
void scan_kernel(const float* __restrict__ xdbl,   // [M,80]
                 const float* __restrict__ dtv,    // [M,DINNER] (softplus'd)
                 const float* __restrict__ xv,     // [M,DINNER] conv+silu
                 const float* __restrict__ xz,     // [M,2*DINNER] (z part)
                 const float* __restrict__ Alog,   // [DINNER,16]
                 const float* __restrict__ Dw,     // [DINNER]
                 unsigned short* __restrict__ ybf) // [M,DINNER] bf16
{
  __shared__ float Bs[DSTATE], Cs[DSTATE];
  const int b = blockIdx.x / 3;
  const int d = (blockIdx.x % 3) * 512 + threadIdx.x;

  float A[DSTATE], hst[DSTATE];
#pragma unroll
  for (int s = 0; s < DSTATE; s++) {
    A[s] = -__expf(Alog[d * DSTATE + s]);
    hst[s] = 0.f;
  }
  const float Dd = Dw[d];

  for (int t = 0; t < SEQ; t++) {
    const long long m = (long long)b * SEQ + t;
    __syncthreads();                 // protect Bs/Cs from previous iteration
    if (threadIdx.x < 32) {
      float v = xdbl[m * 80 + DTRANK + threadIdx.x];
      if (threadIdx.x < DSTATE) Bs[threadIdx.x] = v;
      else                      Cs[threadIdx.x - DSTATE] = v;
    }
    __syncthreads();

    const float dt = dtv[m * DINNER + d];
    const float x  = xv [m * DINNER + d];
    const float z  = xz [m * (2 * DINNER) + DINNER + d];
    const float dx = dt * x;
    float y = 0.f;
#pragma unroll
    for (int s = 0; s < DSTATE; s++) {
      hst[s] = __expf(dt * A[s]) * hst[s] + dx * Bs[s];
      y += hst[s] * Cs[s];
    }
    ybf[m * DINNER + d] = f2bf((y + x * Dd) * siluf(z));
  }
}

// Output head (N=2): scatter value_outputs (t<256) then eos_logits.
__global__ void head_kernel(const float* __restrict__ h,
                            const float* __restrict__ ow,
                            const float* __restrict__ ob,
                            float* __restrict__ out)
{
  int m = blockIdx.x * blockDim.x + threadIdx.x;
  if (m >= MROWS) return;
  const float* hr = h + (size_t)m * DMODEL;
  float s0 = ob[0], s1 = ob[1];
  for (int k = 0; k < DMODEL; k++) {
    float v = hr[k];
    s0 += v * ow[k];
    s1 += v * ow[DMODEL + k];
  }
  int b = m / SEQ, t = m % SEQ;
  if (t < SEQ - 1) out[b * (SEQ - 1) + t] = s0;            // value_outputs
  out[BATCH * (SEQ - 1) + b * SEQ + t] = s1;               // eos_logits
}

// ---------------------------------------------------------------------------
extern "C" void kernel_launch(void* const* d_in, const int* in_sizes, int n_in,
                              void* d_out, int out_size, void* d_ws, size_t ws_size,
                              hipStream_t stream)
{
  (void)in_sizes; (void)n_in; (void)out_size; (void)ws_size;
  const float* physical   = (const float*)d_in[0];
  const float* target_seq = (const float*)d_in[1];
  const float* phys_w     = (const float*)d_in[2];
  const float* phys_b     = (const float*)d_in[3];
  const float* val_w      = (const float*)d_in[4];
  const float* val_b      = (const float*)d_in[5];
  const float* start_tok  = (const float*)d_in[6];
  const float* in_proj_w  = (const float*)d_in[7];
  const float* conv_w     = (const float*)d_in[8];
  const float* conv_b     = (const float*)d_in[9];
  const float* x_proj_w   = (const float*)d_in[10];
  const float* dt_proj_w  = (const float*)d_in[11];
  const float* dt_proj_b  = (const float*)d_in[12];
  const float* A_log      = (const float*)d_in[13];
  const float* Dw         = (const float*)d_in[14];
  const float* out_proj_w = (const float*)d_in[15];
  const float* outp_w     = (const float*)d_in[16];
  const float* outp_b     = (const float*)d_in[17];

  // Workspace carve (256B aligned regions).
  char* p = (char*)d_ws;
  auto carve = [&](size_t bytes) -> void* {
    void* r = (void*)p;
    p += (bytes + 255) & ~(size_t)255;
    return r;
  };
  float*          cond  = (float*)carve((size_t)BATCH * DMODEL * 4);
  float*          hbuf  = (float*)carve((size_t)MROWS * DMODEL * 4);
  unsigned short* hbf   = (unsigned short*)carve((size_t)MROWS * DMODEL * 2);
  unsigned short* wbf   = (unsigned short*)carve((size_t)2 * DINNER * DMODEL * 2);
  float*          xz    = (float*)carve((size_t)MROWS * 2 * DINNER * 4);
  float*          xc    = (float*)carve((size_t)MROWS * DINNER * 4);
  unsigned short* xcbf  = (unsigned short*)carve((size_t)MROWS * DINNER * 2);
  float*          xdbl  = (float*)carve((size_t)MROWS * 80 * 4);
  unsigned short* dtin  = (unsigned short*)carve((size_t)MROWS * DTPAD * 2);
  float*          dtval = (float*)carve((size_t)MROWS * DINNER * 4);
  unsigned short* ybf   = (unsigned short*)carve((size_t)MROWS * DINNER * 2);

  const dim3 T256(256);
  const long long nh = (long long)MROWS * DMODEL;

  cond_kernel<<<(BATCH * DMODEL + 255) / 256, T256, 0, stream>>>(
      physical, phys_w, phys_b, cond);
  embed_kernel<<<(unsigned)((nh + 255) / 256), T256, 0, stream>>>(
      target_seq, val_w, val_b, start_tok, cond, hbuf, hbf);

  for (int i = 0; i < NLAYERS; i++) {
    // ---- in_proj: [M,768] x [3072,768]^T -> xz [M,3072]
    const long long nw = (long long)2 * DINNER * DMODEL;
    f32_to_bf16_kernel<<<(unsigned)((nw + 255) / 256), T256, 0, stream>>>(
        in_proj_w + (size_t)i * nw, wbf, nw);
    gemm_bf16_wmma<2, 4, 2, 4><<<dim3(MROWS / 64, (2 * DINNER) / 256), dim3(256), 0, stream>>>(
        hbf, wbf, xz, MROWS, 2 * DINNER, DMODEL, nullptr, 0);

    // ---- depthwise conv + SiLU -> xc (f32) + xcbf (bf16)
    const long long nc = (long long)MROWS * DINNER;
    conv_silu_kernel<<<(unsigned)((nc + 255) / 256), T256, 0, stream>>>(
        xz, conv_w + (size_t)i * DINNER * 4, conv_b + (size_t)i * DINNER, xc, xcbf);

    // ---- x_proj: [M,1536] x [80,1536]^T -> xdbl [M,80]
    const long long nx = (long long)80 * DINNER;
    f32_to_bf16_kernel<<<(unsigned)((nx + 255) / 256), T256, 0, stream>>>(
        x_proj_w + (size_t)i * nx, wbf, nx);
    gemm_bf16_wmma<4, 1, 1, 1><<<dim3(MROWS / 64, 80 / 16), dim3(128), 0, stream>>>(
        xcbf, wbf, xdbl, MROWS, 80, DINNER, nullptr, 0);

    // ---- dt_proj: pad K 48->64, bias + softplus fused in GEMM epilogue
    const long long nd = (long long)MROWS * DTPAD;
    padk_bf16_kernel<<<(unsigned)((nd + 255) / 256), T256, 0, stream>>>(
        xdbl, dtin, MROWS, DTRANK, 80, DTPAD);
    const long long ndw = (long long)DINNER * DTPAD;
    padk_bf16_kernel<<<(unsigned)((ndw + 255) / 256), T256, 0, stream>>>(
        dt_proj_w + (size_t)i * DINNER * DTRANK, wbf, DINNER, DTRANK, DTRANK, DTPAD);
    gemm_bf16_wmma<2, 4, 2, 4><<<dim3(MROWS / 64, DINNER / 256), dim3(256), 0, stream>>>(
        dtin, wbf, dtval, MROWS, DINNER, DTPAD, dt_proj_b + (size_t)i * DINNER, 1);

    // ---- selective scan (fused D-skip + SiLU gate) -> ybf (bf16)
    scan_kernel<<<dim3(BATCH * 3), dim3(512), 0, stream>>>(
        xdbl, dtval, xc, xz,
        A_log + (size_t)i * DINNER * DSTATE, Dw + (size_t)i * DINNER, ybf);

    // ---- out_proj: [M,1536] x [768,1536]^T -> hbuf [M,768]
    const long long no = (long long)DMODEL * DINNER;
    f32_to_bf16_kernel<<<(unsigned)((no + 255) / 256), T256, 0, stream>>>(
        out_proj_w + (size_t)i * no, wbf, no);
    gemm_bf16_wmma<2, 4, 2, 4><<<dim3(MROWS / 64, DMODEL / 256), dim3(256), 0, stream>>>(
        ybf, wbf, hbuf, MROWS, DMODEL, DINNER, nullptr, 0);
    f32_to_bf16_kernel<<<(unsigned)((nh + 255) / 256), T256, 0, stream>>>(hbuf, hbf, nh);
  }

  head_kernel<<<(MROWS + 255) / 256, T256, 0, stream>>>(
      hbuf, outp_w, outp_b, (float*)d_out);
}
